// EcosystemTransformer_52956946759755
// MI455X (gfx1250) — matrix operand order repaired and checked by reading, not compiled, not run on hardware
//
#include <hip/hip_runtime.h>
#include <hip/hip_bf16.h>

#define B_ 4
#define N_ 1024
#define D_ 512
#define H_ 8
#define DH_ 64
#define L_ 6
#define DFF_ 2048
#define M_ (B_*N_)
#define NEG_INF_ -1000000000.0f

typedef __attribute__((ext_vector_type(16))) _Float16 v16h;
typedef __attribute__((ext_vector_type(8)))  float    v8f;
typedef __attribute__((ext_vector_type(4)))  unsigned int uint4v;

union Frag { v16h h; uint4v u[2]; };

// A-matrix 16x32 f16 fragment: lane (g = lane>>4) holds K = 8g..8g+7 (elems 0..7)
// and K = 16+8g..16+8g+7 (elems 8..15); row m = lane&15 (per ISA 7.12.2).
__device__ inline v16h load_a_frag(const _Float16* rowp, int k0, int g) {
  Frag f;
  f.u[0] = *(const uint4v*)(rowp + k0 + 8*g);
  f.u[1] = *(const uint4v*)(rowp + k0 + 16 + 8*g);
  return f.h;
}
// B-matrix 32x16 f16 fragment: lane holds K = 8g..8g+7 (elems 0..7) and K = 16+8g..16+8g+7 (elems 8..15); col n = lane&15.
__device__ inline v16h load_b_frag(const _Float16* colp, int k0, int g) {
  Frag f;
  f.u[0] = *(const uint4v*)(colp + k0 + 8*g);
  f.u[1] = *(const uint4v*)(colp + k0 + 16 + 8*g);
  return f.h;
}

__device__ inline v8f wmma32(v16h a, v16h b, v8f c) {
  return __builtin_amdgcn_wmma_f32_16x16x32_f16(false, a, false, b, (short)0, c, false, false);
}

// ---------------- bias precompute: (B,N,N) additive mask ----------------
__global__ __launch_bounds__(256) void bias_kernel(
    const float* __restrict__ xy, const int* __restrict__ alive,
    const int* __restrict__ species, float* __restrict__ bias)
{
  size_t idx = (size_t)blockIdx.x * 256 + threadIdx.x;
  int j = (int)(idx & (N_ - 1));
  int i = (int)((idx >> 10) & (N_ - 1));
  int b = (int)(idx >> 20);
  float xi = xy[(size_t)(b*N_ + i)*2], yi = xy[(size_t)(b*N_ + i)*2 + 1];
  float xj = xy[(size_t)(b*N_ + j)*2], yj = xy[(size_t)(b*N_ + j)*2 + 1];
  float dx = xi - xj, dy = yi - yj;
  float d2 = dx*dx + dy*dy;
  float v = 0.0f;
  if (!alive[b*N_ + j]) v += NEG_INF_;
  if (d2 > 100.0f)      v += NEG_INF_;
  int si = species[b*N_ + i], sj = species[b*N_ + j];
  bool ia = si < 2, ip = si == 2, ja = sj < 2, jp = sj == 2;
  if ((ia && jp) || (ip && ja)) v += NEG_INF_;
  bias[idx] = v;
}

// ---------------- LayerNorm: one wave per token (D=512 -> 16 f32/lane) ----------------
__global__ __launch_bounds__(256) void ln_kernel(
    const float* __restrict__ x, const float* __restrict__ gg,
    const float* __restrict__ bb, _Float16* __restrict__ outH,
    float* __restrict__ outF)
{
  const int tok = blockIdx.x * 8 + (threadIdx.x >> 5);
  const int l = threadIdx.x & 31;
  const float* xr = x + (size_t)tok * D_;
  float s = 0.0f, ss = 0.0f;
  float vals[16];
#pragma unroll
  for (int t = 0; t < 16; t++) {
    float v = xr[l + 32*t];
    vals[t] = v; s += v; ss += v*v;
  }
  for (int off = 1; off < 32; off <<= 1) {
    s  += __shfl_xor(s,  off, 32);
    ss += __shfl_xor(ss, off, 32);
  }
  float mean = s * (1.0f / D_);
  float var  = ss * (1.0f / D_) - mean*mean;
  float rs = rsqrtf(var + 1e-5f);
#pragma unroll
  for (int t = 0; t < 16; t++) {
    int d = l + 32*t;
    float v = (vals[t] - mean) * rs * gg[d] + bb[d];
    if (outF) outF[(size_t)tok*D_ + d] = v;
    else      outH[(size_t)tok*D_ + d] = (_Float16)v;
  }
}

// ---------------- weight convert+transpose f32[K][Nc] -> f16 W^T[Nc][K] ----------------
// 32x32 LDS tile transpose: coalesced f32 reads (n fast), coalesced f16 writes (k fast).
// [32][33] pad: transposed read stride 33 words, gcd(33,64)=1 -> conflict-free on 64 banks.
__global__ __launch_bounds__(256) void wconv_kernel(
    const float* __restrict__ W, _Float16* __restrict__ wT, int K, int Nc)
{
  __shared__ float tile[32][33];
  const int tx = threadIdx.x & 31;
  const int ty = threadIdx.x >> 5;          // 8 rows per pass
  const int ntn = Nc >> 5;
  const int tk = blockIdx.x / ntn;
  const int tn = blockIdx.x - tk * ntn;
  const int k0 = tk * 32, n0 = tn * 32;
#pragma unroll
  for (int r = 0; r < 32; r += 8)
    tile[ty + r][tx] = W[(size_t)(k0 + ty + r) * Nc + n0 + tx];
  __syncthreads();
#pragma unroll
  for (int r = 0; r < 32; r += 8)
    wT[(size_t)(n0 + ty + r) * K + k0 + tx] = (_Float16)tile[tx][ty + r];
}

// ---------------- V transpose: [b][j][h*64+d] -> vT[b][h][d][j] ----------------
__global__ __launch_bounds__(256) void vtrans_kernel(
    const _Float16* __restrict__ v16, _Float16* __restrict__ vT)
{
  size_t idx = (size_t)blockIdx.x * 256 + threadIdx.x;  // over B*H*DH*N
  int j = (int)(idx & (N_ - 1));
  int d = (int)((idx >> 10) & (DH_ - 1));
  int h = (int)((idx >> 16) & (H_ - 1));
  int b = (int)(idx >> 19);
  vT[idx] = v16[(size_t)(b*N_ + j)*D_ + h*DH_ + d];
}

// ---------------- WMMA GEMM: C[M][Nc] = A16[M][K] * BT16[Nc][K]^T (+bias/gelu/res) ----------------
#define GF_BIAS  1
#define GF_GELU  2
#define GF_RES   4
#define GF_OUT16 8

__global__ __launch_bounds__(256) void gemm_kernel(
    const _Float16* __restrict__ A, const _Float16* __restrict__ BT,
    const float* __restrict__ bias, const float* __restrict__ res,
    float* __restrict__ outF, _Float16* __restrict__ outH,
    int K, int Nc, int flags)
{
  const int tid = threadIdx.x;
  const int l = tid & 31, lm = l & 15, g = l >> 4;
  const int wid = tid >> 5;
  const int wm = wid >> 2, wn = wid & 3;
  const int rbase = blockIdx.y * 64 + wm * 32;
  const int cbase = blockIdx.x * 128 + wn * 32;

  const _Float16* arow0 = A  + (size_t)(rbase + lm)      * K;
  const _Float16* arow1 = A  + (size_t)(rbase + 16 + lm) * K;
  const _Float16* bcol0 = BT + (size_t)(cbase + lm)      * K;
  const _Float16* bcol1 = BT + (size_t)(cbase + 16 + lm) * K;

  v8f acc00 = {}, acc01 = {}, acc10 = {}, acc11 = {};
  for (int k0 = 0; k0 < K; k0 += 32) {
    // pull the streams ~4 k-steps ahead into WGP$ (global_prefetch_b8, no LOADcnt)
    if (k0 + 128 < K) {
      __builtin_prefetch(arow0 + k0 + 128, 0, 3);
      __builtin_prefetch(arow1 + k0 + 128, 0, 3);
      __builtin_prefetch(bcol0 + k0 + 128, 0, 3);
      __builtin_prefetch(bcol1 + k0 + 128, 0, 3);
    }
    v16h a0 = load_a_frag(arow0, k0, g);
    v16h a1 = load_a_frag(arow1, k0, g);
    v16h b0 = load_b_frag(bcol0, k0, g);
    v16h b1 = load_b_frag(bcol1, k0, g);
    acc00 = wmma32(a0, b0, acc00);
    acc01 = wmma32(a0, b1, acc01);
    acc10 = wmma32(a1, b0, acc10);
    acc11 = wmma32(a1, b1, acc11);
  }

  v8f accs[2][2] = { {acc00, acc01}, {acc10, acc11} };
#pragma unroll
  for (int i = 0; i < 2; i++) {
#pragma unroll
    for (int j = 0; j < 2; j++) {
      int col = cbase + j*16 + lm;
      float bv = (flags & GF_BIAS) ? bias[col] : 0.0f;
#pragma unroll
      for (int r = 0; r < 8; r++) {
        int row = rbase + i*16 + r + 8*g;
        float v = accs[i][j][r] + bv;
        if (flags & GF_GELU) v = 0.5f * v * (1.0f + erff(v * 0.70710678f));
        if (flags & GF_RES)  v += res[(size_t)row * Nc + col];
        if (flags & GF_OUT16) outH[(size_t)row * Nc + col] = (_Float16)v;
        else                  outF[(size_t)row * Nc + col] = v;
      }
    }
  }
}

// ---------------- flash attention: one wave per (b, h, 16-row Q tile) ----------------
__global__ __launch_bounds__(256) void attn_kernel(
    const _Float16* __restrict__ q16, const _Float16* __restrict__ k16,
    const _Float16* __restrict__ vT16, const float* __restrict__ bias,
    _Float16* __restrict__ o16)
{
  __shared__ alignas(16) _Float16 Pb[8][16][32];   // per-wave P staging (A-layout re-entry)
  const int tid = threadIdx.x;
  const int l = tid & 31, lm = l & 15, g = l >> 4;
  const int wl = tid >> 5;
  int wid = blockIdx.x * 8 + wl;
  const int mt = wid & (N_/16 - 1); wid >>= 6;
  const int h  = wid & (H_ - 1);    wid >>= 3;
  const int b  = wid;
  const int m0 = mt * 16;

  const _Float16* qrow = q16 + ((size_t)(b*N_ + m0 + lm) * D_ + h*DH_);
  v16h aq0 = load_a_frag(qrow, 0,  g);
  v16h aq1 = load_a_frag(qrow, 32, g);

  v8f acc0 = {}, acc1 = {}, acc2 = {}, acc3 = {};
  float mrow[8], srow[8];
#pragma unroll
  for (int r = 0; r < 8; r++) { mrow[r] = -INFINITY; srow[r] = 0.0f; }

  const float* biasb = bias + (size_t)b * N_ * N_;

  for (int j0 = 0; j0 < N_; j0 += 32) {
    // S = Q K^T / 8 + bias  (two 16x16 tiles)
    v8f s[2];
#pragma unroll
    for (int jt = 0; jt < 2; jt++) {
      const _Float16* krow = k16 + ((size_t)(b*N_ + j0 + jt*16 + lm) * D_ + h*DH_);
      v16h bk0 = load_b_frag(krow, 0,  g);
      v16h bk1 = load_b_frag(krow, 32, g);
      v8f sv = {};
      sv = wmma32(aq0, bk0, sv);
      sv = wmma32(aq1, bk1, sv);
      s[jt] = sv;
    }
#pragma unroll
    for (int jt = 0; jt < 2; jt++) {
#pragma unroll
      for (int r = 0; r < 8; r++) {
        int i = m0 + r + 8*g;
        int j = j0 + jt*16 + lm;
        s[jt][r] = s[jt][r] * 0.125f + biasb[(size_t)i * N_ + j];
      }
    }
    // online softmax: row max across 16 lanes (rows live at fixed lane-group)
    float f[8];
#pragma unroll
    for (int r = 0; r < 8; r++) {
      float v = fmaxf(s[0][r], s[1][r]);
      for (int off = 1; off < 16; off <<= 1) v = fmaxf(v, __shfl_xor(v, off, 32));
      float nm = fmaxf(mrow[r], v);
      f[r] = __expf(mrow[r] - nm);
      mrow[r] = nm;
    }
    float ps[8];
#pragma unroll
    for (int r = 0; r < 8; r++) ps[r] = 0.0f;
#pragma unroll
    for (int jt = 0; jt < 2; jt++) {
#pragma unroll
      for (int r = 0; r < 8; r++) {
        float p = __expf(s[jt][r] - mrow[r]);
        ps[r] += p;
        Pb[wl][r + 8*g][jt*16 + lm] = (_Float16)p;
      }
    }
#pragma unroll
    for (int r = 0; r < 8; r++) {
      float v = ps[r];
      for (int off = 1; off < 16; off <<= 1) v += __shfl_xor(v, off, 32);
      srow[r] = srow[r] * f[r] + v;
    }
#pragma unroll
    for (int r = 0; r < 8; r++) {
      acc0[r] *= f[r]; acc1[r] *= f[r]; acc2[r] *= f[r]; acc3[r] *= f[r];
    }

    asm volatile("s_wait_dscnt 0" ::: "memory");  // P stores visible before A-layout reload

    Frag pf;
    pf.u[0] = *(const uint4v*)&Pb[wl][lm][8*g];
    pf.u[1] = *(const uint4v*)&Pb[wl][lm][16 + 8*g];
    v16h pa = pf.h;

    const _Float16* vb = vT16 + ((size_t)((b*H_ + h) * DH_ + lm) * N_ + j0);
    {
      v16h bv0 = load_b_frag(vb + (size_t)0  * N_, 0, g);
      v16h bv1 = load_b_frag(vb + (size_t)16 * N_, 0, g);
      v16h bv2 = load_b_frag(vb + (size_t)32 * N_, 0, g);
      v16h bv3 = load_b_frag(vb + (size_t)48 * N_, 0, g);
      acc0 = wmma32(pa, bv0, acc0);
      acc1 = wmma32(pa, bv1, acc1);
      acc2 = wmma32(pa, bv2, acc2);
      acc3 = wmma32(pa, bv3, acc3);
    }
  }

#pragma unroll
  for (int r = 0; r < 8; r++) {
    int row = m0 + r + 8*g;
    float inv = 1.0f / srow[r];
    size_t base = (size_t)(b*N_ + row) * D_ + h*DH_ + lm;
    o16[base + 0 ] = (_Float16)(acc0[r] * inv);
    o16[base + 16] = (_Float16)(acc1[r] * inv);
    o16[base + 32] = (_Float16)(acc2[r] * inv);
    o16[base + 48] = (_Float16)(acc3[r] * inv);
  }
}

// ---------------- host orchestration ----------------
extern "C" void kernel_launch(void* const* d_in, const int* in_sizes, int n_in,
                              void* d_out, int out_size, void* d_ws, size_t ws_size,
                              hipStream_t stream) {
  const float* tokens = (const float*)d_in[0];
  const float* xy     = (const float*)d_in[1];
  const float* Wq     = (const float*)d_in[2];
  const float* Wk     = (const float*)d_in[3];
  const float* Wv     = (const float*)d_in[4];
  const float* Wo     = (const float*)d_in[5];
  const float* bo     = (const float*)d_in[6];
  const float* W1     = (const float*)d_in[7];
  const float* b1     = (const float*)d_in[8];
  const float* W2     = (const float*)d_in[9];
  const float* b2     = (const float*)d_in[10];
  const float* g1     = (const float*)d_in[11];
  const float* be1    = (const float*)d_in[12];
  const float* g2     = (const float*)d_in[13];
  const float* be2    = (const float*)d_in[14];
  const float* gf     = (const float*)d_in[15];
  const float* bf     = (const float*)d_in[16];
  const int*   alive   = (const int*)d_in[17];
  const int*   species = (const int*)d_in[18];
  (void)in_sizes; (void)n_in; (void)out_size; (void)ws_size;

  char* p = (char*)d_ws;
  auto carve = [&](size_t bytes) { void* r = (void*)p; p += (bytes + 255) & ~(size_t)255; return r; };
  float*    x    = (float*)   carve((size_t)M_*D_*4);
  _Float16* h16  = (_Float16*)carve((size_t)M_*D_*2);
  _Float16* q16  = (_Float16*)carve((size_t)M_*D_*2);
  _Float16* k16  = (_Float16*)carve((size_t)M_*D_*2);
  _Float16* v16  = (_Float16*)carve((size_t)M_*D_*2);
  _Float16* vT16 = (_Float16*)carve((size_t)M_*D_*2);
  _Float16* o16  = (_Float16*)carve((size_t)M_*D_*2);
  _Float16* a16  = (_Float16*)carve((size_t)M_*DFF_*2);
  float*    bias = (float*)   carve((size_t)B_*N_*N_*4);
  _Float16* wT   = (_Float16*)carve((size_t)D_*DFF_*2);

  hipMemcpyAsync(x, tokens, (size_t)M_*D_*4, hipMemcpyDeviceToDevice, stream);
  bias_kernel<<<(B_*N_*N_)/256, 256, 0, stream>>>(xy, alive, species, bias);

  for (int l = 0; l < L_; l++) {
    ln_kernel<<<M_/8, 256, 0, stream>>>(x, g1 + (size_t)l*D_, be1 + (size_t)l*D_, h16, nullptr);

    wconv_kernel<<<(D_/32)*(D_/32), 256, 0, stream>>>(Wq + (size_t)l*D_*D_, wT, D_, D_);
    gemm_kernel<<<dim3(D_/128, M_/64), 256, 0, stream>>>(h16, wT, nullptr, nullptr, nullptr, q16, D_, D_, GF_OUT16);
    wconv_kernel<<<(D_/32)*(D_/32), 256, 0, stream>>>(Wk + (size_t)l*D_*D_, wT, D_, D_);
    gemm_kernel<<<dim3(D_/128, M_/64), 256, 0, stream>>>(h16, wT, nullptr, nullptr, nullptr, k16, D_, D_, GF_OUT16);
    wconv_kernel<<<(D_/32)*(D_/32), 256, 0, stream>>>(Wv + (size_t)l*D_*D_, wT, D_, D_);
    gemm_kernel<<<dim3(D_/128, M_/64), 256, 0, stream>>>(h16, wT, nullptr, nullptr, nullptr, v16, D_, D_, GF_OUT16);

    vtrans_kernel<<<(M_*D_)/256, 256, 0, stream>>>(v16, vT16);
    attn_kernel<<<(B_*H_*(N_/16))/8, 256, 0, stream>>>(q16, k16, vT16, bias, o16);

    wconv_kernel<<<(D_/32)*(D_/32), 256, 0, stream>>>(Wo + (size_t)l*D_*D_, wT, D_, D_);
    gemm_kernel<<<dim3(D_/128, M_/64), 256, 0, stream>>>(o16, wT, bo + (size_t)l*D_, x, x, nullptr, D_, D_, GF_BIAS | GF_RES);

    ln_kernel<<<M_/8, 256, 0, stream>>>(x, g2 + (size_t)l*D_, be2 + (size_t)l*D_, h16, nullptr);

    wconv_kernel<<<(D_/32)*(DFF_/32), 256, 0, stream>>>(W1 + (size_t)l*D_*DFF_, wT, D_, DFF_);
    gemm_kernel<<<dim3(DFF_/128, M_/64), 256, 0, stream>>>(h16, wT, b1 + (size_t)l*DFF_, nullptr, nullptr, a16, D_, DFF_, GF_BIAS | GF_GELU | GF_OUT16);

    wconv_kernel<<<(DFF_/32)*(D_/32), 256, 0, stream>>>(W2 + (size_t)l*DFF_*D_, wT, DFF_, D_);
    gemm_kernel<<<dim3(D_/128, M_/64), 256, 0, stream>>>(a16, wT, b2 + (size_t)l*D_, x, x, nullptr, DFF_, D_, GF_BIAS | GF_RES);
  }

  ln_kernel<<<M_/8, 256, 0, stream>>>(x, gf, bf, nullptr, (float*)d_out);
}
